// SelfAttention_89343909692120
// MI455X (gfx1250) — matrix operand order repaired and checked by reading, not compiled, not run on hardware
//
#include <hip/hip_runtime.h>

#define B_   8
#define S_   1024
#define D_   768
#define SS_  32          // 32x32 spatial grid
#define H_   12
#define HD_  64
#define EPS_ 1e-7f

typedef __attribute__((ext_vector_type(8)))  float  v8f;
typedef __attribute__((ext_vector_type(8)))  __bf16 v8bf;
typedef __attribute__((ext_vector_type(16))) __bf16 v16bf;
typedef __attribute__((ext_vector_type(4)))  unsigned int v4u;
typedef __attribute__((ext_vector_type(8)))  int v8i;
typedef __attribute__((ext_vector_type(4)))  int v4i;
typedef unsigned short u16;

// TDM availability (device pass only; host pass parses the fallback branch)
#if defined(__HIP_DEVICE_COMPILE__) && __has_builtin(__builtin_amdgcn_tensor_load_to_lds)
#define HAVE_TDM 1
#else
#define HAVE_TDM 0
#endif

// Padded B-panel geometry in LDS: 64 rows x 768 bf16, +1 dword pad per 64 dwords
// row stride = 384 + 6 = 390 dwords = 780 u16 = 1560 B ; stride%64banks = 6 -> conflict-free b128 reads
#define BP_ROW_U16 780
#define BP_BYTES   (64 * BP_ROW_U16 * 2)   // 99840

// ---------- helpers ----------
__device__ __forceinline__ u16 f2bf(float f) {
    union { float f; unsigned u; } v; v.f = f;
    unsigned r = v.u + 0x7FFFu + ((v.u >> 16) & 1u);   // round-to-nearest-even
    return (u16)(r >> 16);
}

// A-matrix fragment (16x32 bf16): p = row_base + ks*32 + lhalf*16.
// elements 0..15 = K kb..kb+15 (16 contiguous K values per lane)
__device__ __forceinline__ v16bf frag_a(const u16* p) {
    const v8bf* q = (const v8bf*)p;
    v8bf lo = q[0], hi = q[1];
    return __builtin_shufflevector(lo, hi, 0,1,2,3,4,5,6,7,8,9,10,11,12,13,14,15);
}
// B-matrix fragment (32x16 bf16) from B^T rows (N-major, K contiguous):
// p = nrow_base + ks*32 + lhalf*16 ; 16 contiguous K values per lane.
__device__ __forceinline__ v16bf frag_b(const u16* p) {
    const v8bf* q = (const v8bf*)p;
    v8bf lo = q[0], hi = q[1];
    return __builtin_shufflevector(lo, hi, 0,1,2,3,4,5,6,7,8,9,10,11,12,13,14,15);
}
__device__ __forceinline__ v8f wmma_bf16(v16bf a, v16bf b, v8f c) {
    return __builtin_amdgcn_wmma_f32_16x16x32_bf16(false, a, false, b, (short)0, c, false, false);
}

// ---------- kernel 1: fp32 -> bf16 convert ----------
__global__ void k_f2bf(const float* __restrict__ src, u16* __restrict__ dst, int n) {
    int i = blockIdx.x * 256 + threadIdx.x;
    if (i < n) dst[i] = f2bf(src[i]);
}

// ---------- kernel 2: horizontal 7-tap sum of x^2 ----------
__global__ void k_rms_rowsum(const float* __restrict__ x, float* __restrict__ rowsum) {
    int idx = blockIdx.x * 256 + threadIdx.x;
    if (idx >= B_ * SS_ * SS_ * D_) return;
    int d = idx % D_;
    int t = idx / D_;
    int j = t % SS_;  t /= SS_;
    int i = t % SS_;
    int b = t / SS_;
    float s = 0.f;
    #pragma unroll
    for (int dj = -3; dj <= 3; ++dj) {
        int jj = j + dj;
        if (jj >= 0 && jj < SS_) {
            float v = x[((size_t)(b * SS_ + i) * SS_ + jj) * D_ + d];
            s += v * v;
        }
    }
    rowsum[idx] = s;
}

// ---------- kernel 3: vertical 7-tap sum + rms apply -> bf16 xn ----------
__global__ void k_rms_apply(const float* __restrict__ x, const float* __restrict__ rowsum,
                            const float* __restrict__ w, u16* __restrict__ xn) {
    int idx = blockIdx.x * 256 + threadIdx.x;
    if (idx >= B_ * SS_ * SS_ * D_) return;
    int d = idx % D_;
    int t = idx / D_;
    int j = t % SS_;  t /= SS_;
    int i = t % SS_;
    int b = t / SS_;
    float s = 0.f;
    #pragma unroll
    for (int di = -3; di <= 3; ++di) {
        int ii = i + di;
        if (ii >= 0 && ii < SS_)
            s += rowsum[((size_t)(b * SS_ + ii) * SS_ + j) * D_ + d];
    }
    float rms = sqrtf(EPS_ + s * (1.0f / 49.0f));
    float xv  = x[idx];
    xn[idx] = f2bf(xv / rms * w[d]);
}

// ---------- kernel 4: GEMM  C[M,N] = A[M,K] * Wt[N,K]^T + bias ----------
// A: bf16 [8192, 768] row-major. Wbf: bf16 [N=768, K=768] row-major (acts as B^T).
// B panel (64 rows x 768) staged to LDS once per block via TDM (padded layout).
// mode 0: bf16 out per-head [b,h,s,hd]; mode 1: bf16 out per-head [b,h,hd,s] (V^T);
// mode 2: fp32 out [M,N].
__global__ void k_proj(const u16* __restrict__ A, const u16* __restrict__ Wbf,
                       const float* __restrict__ bias, void* __restrict__ out,
                       int mode, float scale) {
    extern __shared__ u16 bsm[];  // padded B panel
    int nb = blockIdx.x;          // 0..11   (N/64)
    int mb = blockIdx.y;          // 0..127  (M/64)
    int tid = threadIdx.x;
    int wave = tid >> 5, lane = tid & 31;
    int l15 = lane & 15, lhalf = (lane >> 4) & 1;

    // ---- stage B panel: rows [nb*64, nb*64+64), all 768 K, into padded LDS ----
#if HAVE_TDM
    if (wave == 0) {
        unsigned long long ga = (unsigned long long)(size_t)(Wbf + (size_t)nb * 64 * D_);
        // D# group0: count=1 | lds_addr=0 | global_addr | type=2
        v4u g0 = { 1u, 0u,
                   (unsigned)(ga & 0xFFFFFFFFu),
                   (unsigned)((ga >> 32) & 0x01FFFFFFu) | 0x80000000u };
        // D# group1: data_size=2B | pad_enable, interval=64dw(code5), amount=1dw(code0)
        //            tensor_dim0=768, tensor_dim1=768, tile_dim0=768, tile_dim1=64,
        //            tensor_dim0_stride=768
        v8i g1 = { (int)0x01510000, (int)0x03000000, (int)0x03000000, (int)0x03000000,
                   64, 768, 0, 0 };
        v4i gz = { 0, 0, 0, 0 };
#if __clang_major__ >= 23
        v8i gz8 = { 0, 0, 0, 0, 0, 0, 0, 0 };
        __builtin_amdgcn_tensor_load_to_lds(g0, g1, gz, gz, gz8, 0);
#else
        __builtin_amdgcn_tensor_load_to_lds(g0, g1, gz, gz, 0);
#endif
    }
#if __has_builtin(__builtin_amdgcn_s_wait_tensorcnt)
    __builtin_amdgcn_s_wait_tensorcnt(0);
#endif
    __syncthreads();
#else
    // fallback: cooperative copy into the same padded layout
    for (int c = tid; c < 64 * 96; c += 256) {       // 96 chunks of 8 u16 per row
        int r = c / 96, col8 = c % 96;
        int cdw = col8 * 4;                          // dword column
        int cup = (cdw + (cdw >> 6)) * 2;            // padded u16 offset
        *(v8bf*)(bsm + (size_t)r * BP_ROW_U16 + cup) =
            *(const v8bf*)(Wbf + (size_t)(nb * 64 + r) * D_ + col8 * 8);
    }
    __syncthreads();
#endif

    int mt  = wave & 3;           // 4 M-tiles
    int ntp = wave >> 2;          // 2 N-tile pairs
    int m0  = mb * 64 + mt * 16;
    int lr0 = ntp * 32 + l15;     // local B row for u=0 (u=1: +16)

    v8f acc[2] = {};
    const u16* abase = A + (size_t)(m0 + l15) * D_ + lhalf * 16;

    for (int ks = 0; ks < D_ / 32; ++ks) {
        int cdw = ks * 16 + lhalf * 8;               // dword column in B panel
        int cup = (cdw + (cdw >> 6)) * 2;            // padded u16 offset
        v16bf af = frag_a(abase + ks * 32);
        acc[0] = wmma_bf16(af, frag_b(bsm + (size_t)lr0 * BP_ROW_U16 + cup), acc[0]);
        acc[1] = wmma_bf16(af, frag_b(bsm + (size_t)(lr0 + 16) * BP_ROW_U16 + cup), acc[1]);
    }

    #pragma unroll
    for (int u = 0; u < 2; ++u) {
        int n = nb * 64 + (ntp * 2 + u) * 16 + l15;
        float bn = bias[n];
        #pragma unroll
        for (int v = 0; v < 8; ++v) {
            int m = m0 + v + 8 * lhalf;
            float val = (acc[u][v] + bn) * scale;
            if (mode == 2) {
                ((float*)out)[(size_t)m * D_ + n] = val;
            } else {
                int bb = m >> 10, s = m & 1023;
                int hh = n >> 6,  dh = n & 63;
                u16* o = (u16*)out;
                if (mode == 0)
                    o[((size_t)(bb * H_ + hh) * S_ + s) * HD_ + dh] = f2bf(val);
                else
                    o[((size_t)(bb * H_ + hh) * HD_ + dh) * S_ + s] = f2bf(val);
            }
        }
    }
}

// ---------- kernel 5: attention for one (b,h), 32-query slab ----------
// qbuf/kbuf: bf16 [b,h,s,hd]; vbuf: bf16 [b,h,hd,s] (V^T); out: bf16 [b*s, D] tokens-major.
// Dynamic LDS: 32*1024 floats (128KB). Probabilities rewritten in-place as bf16.
__global__ void k_attn(const u16* __restrict__ qbuf, const u16* __restrict__ kbuf,
                       const u16* __restrict__ vbuf, u16* __restrict__ attnb) {
    extern __shared__ float smem[];                 // [32][1024]
    int qblk = blockIdx.x;                          // 0..31
    int bh   = blockIdx.y;                          // 0..95
    int b = bh / H_, hh = bh % H_;
    int tid = threadIdx.x;
    int wave = tid >> 5, lane = tid & 31;
    int l15 = lane & 15, lhalf = (lane >> 4) & 1;
    int qrow0 = qblk * 32;

    const u16* qbase = qbuf + (size_t)bh * S_ * HD_;
    const u16* kbase = kbuf + (size_t)bh * S_ * HD_;
    const u16* vbase = vbuf + (size_t)bh * HD_ * S_;

    // ---- phase 1: scores[32][1024] = (Q/8) @ K^T ----
    v16bf qa[2][2];
    #pragma unroll
    for (int mtq = 0; mtq < 2; ++mtq)
        #pragma unroll
        for (int ks = 0; ks < 2; ++ks)
            qa[mtq][ks] = frag_a(qbase + (size_t)(qrow0 + mtq * 16 + l15) * HD_ + ks * 32 + lhalf * 16);

    for (int t = 0; t < 8; ++t) {
        int n0 = (wave + 8 * t) * 16;
        v8f acc0 = {}, acc1 = {};
        #pragma unroll
        for (int ks = 0; ks < 2; ++ks) {
            v16bf kf = frag_b(kbase + (size_t)(n0 + l15) * HD_ + ks * 32 + lhalf * 16);
            acc0 = wmma_bf16(qa[0][ks], kf, acc0);
            acc1 = wmma_bf16(qa[1][ks], kf, acc1);
        }
        #pragma unroll
        for (int v = 0; v < 8; ++v) {
            int r = v + 8 * lhalf;
            smem[(size_t)r * S_        + n0 + l15] = acc0[v];
            smem[(size_t)(16 + r) * S_ + n0 + l15] = acc1[v];
        }
    }
    __syncthreads();

    // ---- phase 2: exact softmax per row; write bf16 P in-place ----
    #pragma unroll
    for (int rr = 0; rr < 4; ++rr) {
        int r = wave * 4 + rr;
        float vals[32];
        float mx = -3.402823466e38f;
        #pragma unroll
        for (int t = 0; t < 32; ++t) {
            vals[t] = smem[(size_t)r * S_ + lane + 32 * t];
            mx = fmaxf(mx, vals[t]);
        }
        #pragma unroll
        for (int off = 16; off > 0; off >>= 1) mx = fmaxf(mx, __shfl_xor(mx, off, 32));
        float sum = 0.f;
        #pragma unroll
        for (int t = 0; t < 32; ++t) { vals[t] = __expf(vals[t] - mx); sum += vals[t]; }
        #pragma unroll
        for (int off = 16; off > 0; off >>= 1) sum += __shfl_xor(sum, off, 32);
        float inv = 1.0f / sum;
        asm volatile("" ::: "memory");              // all row reads precede in-place writes
        u16* prow = (u16*)(smem + (size_t)r * S_);  // row r as 1024 bf16 (2KB of 4KB row)
        #pragma unroll
        for (int t = 0; t < 32; ++t) prow[lane + 32 * t] = f2bf(vals[t] * inv);
    }
    __syncthreads();

    // ---- phase 3: out[32][64] = P[32][1024] @ V ;  V^T rows are B^T rows ----
    int mt = wave & 1, nt = wave >> 1;              // 2 x 4 tiles, one per wave
    v8f acc = {};
    const u16* pbase = (const u16*)smem;            // row stride 2048 u16 (4KB)
    const u16* pa = pbase + (size_t)(mt * 16 + l15) * 2048 + lhalf * 16;
    const u16* pb = vbase + (size_t)(nt * 16 + l15) * S_ + lhalf * 16;
    for (int ks = 0; ks < 32; ++ks) {
        v16bf af = frag_a(pa + ks * 32);
        v16bf bf = frag_b(pb + ks * 32);
        acc = wmma_bf16(af, bf, acc);
    }
    #pragma unroll
    for (int v = 0; v < 8; ++v) {
        int row = qrow0 + mt * 16 + v + 8 * lhalf;
        int dh  = nt * 16 + l15;
        attnb[(size_t)(b * S_ + row) * D_ + hh * HD_ + dh] = f2bf(acc[v]);
    }
}

// ---------- orchestration ----------
extern "C" void kernel_launch(void* const* d_in, const int* in_sizes, int n_in,
                              void* d_out, int out_size, void* d_ws, size_t ws_size,
                              hipStream_t stream) {
    const float* x      = (const float*)d_in[0];
    const float* norm_w = (const float*)d_in[1];
    const float* wq = (const float*)d_in[2];
    const float* bq = (const float*)d_in[3];
    const float* wk = (const float*)d_in[4];
    const float* bk = (const float*)d_in[5];
    const float* wv = (const float*)d_in[6];
    const float* bv = (const float*)d_in[7];
    const float* wo = (const float*)d_in[8];
    const float* bo = (const float*)d_in[9];

    const size_t NTOK = (size_t)B_ * S_;            // 8192
    const size_t XN_B = NTOK * D_ * 2;              // 12,582,912
    const size_t W_B  = (size_t)D_ * D_ * 2;        //  1,179,648
    const size_t RS_B = NTOK * D_ * 4;              // 25,165,824

    char* ws = (char*)d_ws;
    u16*   xn  = (u16*)ws;                                   // bf16 xn
    u16*   wqb = (u16*)(ws + XN_B);
    u16*   wkb = (u16*)(ws + XN_B + W_B);
    u16*   wvb = (u16*)(ws + XN_B + 2 * W_B);
    u16*   wob = (u16*)(ws + XN_B + 3 * W_B);
    float* rsm = (float*)(ws + XN_B + 4 * W_B);              // rowsum (reused below)
    u16*   qb  = (u16*)rsm;                                  // reuse rowsum: Q (12.58MB)
    u16*   kb  = qb + NTOK * HD_ * H_ / 1;                   // == qb + 6,291,456 elems: K
    u16*   vb  = (u16*)(ws + XN_B + 4 * W_B + RS_B);         // V^T
    u16*   ab  = vb + NTOK * D_;                             // attn out bf16 tokens-major

    const int wn = D_ * D_;                                  // 589824
    int wblocks = (wn + 255) / 256;
    k_f2bf<<<wblocks, 256, 0, stream>>>(wq, wqb, wn);
    k_f2bf<<<wblocks, 256, 0, stream>>>(wk, wkb, wn);
    k_f2bf<<<wblocks, 256, 0, stream>>>(wv, wvb, wn);
    k_f2bf<<<wblocks, 256, 0, stream>>>(wo, wob, wn);

    const int en = B_ * SS_ * SS_ * D_;                      // 6,291,456
    int eblocks = (en + 255) / 256;
    k_rms_rowsum<<<eblocks, 256, 0, stream>>>(x, rsm);
    // rsm region is reused as qb/kb only AFTER k_rms_apply has consumed it
    k_rms_apply<<<eblocks, 256, 0, stream>>>(x, rsm, norm_w, xn);

    dim3 pgrid(D_ / 64, (int)(NTOK / 64));                   // (12, 128)
    k_proj<<<pgrid, 256, BP_BYTES, stream>>>(xn, wqb, bq, (void*)qb, 0, 0.125f); // Q/sqrt(HD)
    k_proj<<<pgrid, 256, BP_BYTES, stream>>>(xn, wkb, bk, (void*)kb, 0, 1.0f);   // K
    k_proj<<<pgrid, 256, BP_BYTES, stream>>>(xn, wvb, bv, (void*)vb, 1, 1.0f);   // V^T

    dim3 agrid(S_ / 32, B_ * H_);                            // (32, 96)
    k_attn<<<agrid, 256, 32 * S_ * sizeof(float), stream>>>(qb, kb, vb, ab);

    k_proj<<<pgrid, 256, BP_BYTES, stream>>>(ab, wob, bo, d_out, 2, 1.0f);       // O
}
